// GAT_81329500717147
// MI455X (gfx1250) — matrix operand order repaired and checked by reading, not compiled, not run on hardware
//
#include <hip/hip_runtime.h>
#include <hip/hip_bf16.h>

// ---------------- problem constants ----------------
#define NN      5000
#define NF      512
#define NHID    64
#define NHEADS  8
#define NCLASS  40
#define NPAD    5120          // 40*128, multiple of 32/128 for tiling
#define WPR     (NPAD / 32)   // 160 mask words per row
#define GAT_ALPHA 0.2f

typedef __bf16 bf16_t;
typedef float    v8f   __attribute__((ext_vector_type(8)));
typedef __bf16   v16bf __attribute__((ext_vector_type(16)));
typedef unsigned v4u   __attribute__((ext_vector_type(4)));
typedef int      v4i   __attribute__((ext_vector_type(4)));
typedef int      v8i   __attribute__((ext_vector_type(8)));

#if __has_builtin(__builtin_amdgcn_tensor_load_to_lds) && \
    __has_builtin(__builtin_amdgcn_s_wait_tensorcnt)
#define GAT_USE_TDM 1
#else
#define GAT_USE_TDM 0
#endif

__device__ __forceinline__ v8f zero8() {
  v8f z = {0.f,0.f,0.f,0.f,0.f,0.f,0.f,0.f};
  return z;
}

__device__ __forceinline__ unsigned ballot32(bool p) {
#if __has_builtin(__builtin_amdgcn_ballot_w32)
  return __builtin_amdgcn_ballot_w32(p);
#else
  return (unsigned)__ballot(p ? 1 : 0);
#endif
}

// load a 32-byte (16 x bf16) fragment from two 16B-aligned addresses
__device__ __forceinline__ v16bf ld_frag(const bf16_t* p0, const bf16_t* p1) {
  v16bf f;
  int4* d = reinterpret_cast<int4*>(&f);
  d[0] = *reinterpret_cast<const int4*>(p0);
  d[1] = *reinterpret_cast<const int4*>(p1);
  return f;
}

__device__ __forceinline__ v8f wmma_bf16(v16bf a, v16bf b, v8f c) {
  // D = A(16x32 bf16) * B(32x16 bf16) + C(16x16 f32)
  return __builtin_amdgcn_wmma_f32_16x16x32_bf16(false, a, false, b,
                                                 (short)0, c, false, false);
}

// =====================================================================
// Kernel 0: x (f32, 5000 rows) -> bf16 padded [NPAD x 512] so the GEMM
// A-tiles are a clean 2-D tensor for the Tensor Data Mover (no guards).
// =====================================================================
__global__ void tobf16_kernel(const float* __restrict__ x,
                              bf16_t* __restrict__ xb) {
  int idx = blockIdx.x * blockDim.x + threadIdx.x;
  if (idx >= NPAD * NF) return;
  int r = idx >> 9;
  float v = (r < NN) ? x[idx] : 0.f;
  xb[idx] = (bf16_t)v;
}

// =====================================================================
// Kernel 1: compress adj (100 MB f32, read exactly ONCE) into a bitmask
// (3.2 MB, L2-resident). One block per row, wave32 ballot builds words.
// =====================================================================
__global__ __launch_bounds__(256) void mask_kernel(
    const float* __restrict__ adj, unsigned* __restrict__ mask) {
  const int i    = blockIdx.x;           // padded row
  const int w    = threadIdx.x >> 5;
  const int lane = threadIdx.x & 31;
  for (int wd = w; wd < WPR; wd += 8) {
    unsigned bits = 0;
    if (i < NN) {
      int j = wd * 32 + lane;
      bool pred = (j < NN) && (adj[(size_t)i * NN + j] > 0.f);
      bits = ballot32(pred);
    }
    if (lane == 0) mask[(size_t)i * WPR + wd] = bits;
  }
}

// =====================================================================
// Kernel 2: bf16 WMMA GEMM  A[NPAD x 512](bf16) @ B[head][512 x 64](f32)
// -> outT[head][64][NPAD] (transposed so attention B-fragments are 16
//    contiguous bf16 per lane). A-tile (128x64 bf16) is DMA'd into LDS
//    by the Tensor Data Mover with hardware row padding (+8 bf16/row);
//    wave 0 issues the descriptor and waits on TENSORcnt while all
//    waves stage the transposed B tile.
// =====================================================================
__global__ __launch_bounds__(256) void gemm512_kernel(
    const bf16_t* __restrict__ Abf, const float* __restrict__ Ball,
    bf16_t* __restrict__ outT_all) {
  constexpr int K = 512, TK = 64, BM = 128;
  __shared__ alignas(16) bf16_t As[BM][TK + 8];   // 18.4 KB (72-elem rows)
  __shared__ alignas(16) bf16_t Bs[64][TK + 8];   // 9.2 KB (stored transposed)
  const int head = blockIdx.y;
  const float* Bh = Ball + (size_t)head * K * 64;
  bf16_t* outT = outT_all + (size_t)head * 64 * NPAD;
  const int row0 = blockIdx.x * BM;
  const int tid = threadIdx.x, wave = tid >> 5, lane = tid & 31;
  const int lrow = lane & 15, lhalf = lane >> 4;

  v8f acc[4];
  for (int t = 0; t < 4; ++t) acc[t] = zero8();

  for (int kb = 0; kb < K; kb += TK) {
    __syncthreads();                       // previous tile fully consumed
#if GAT_USE_TDM
    if (wave == 0) {
      // ---- Tensor DMA descriptor (D#): 2-D bf16 tile 64(x) x 128(y) ----
      unsigned long long ga =
          (unsigned long long)(uintptr_t)(Abf + (size_t)row0 * K + kb);
      unsigned ldsa = (unsigned)(uintptr_t)(&As[0][0]);  // LDS byte offset
      v4u g0;
      g0[0] = 1u;                                   // count=1 descriptor
      g0[1] = ldsa;                                 // lds_addr
      g0[2] = (unsigned)ga;                         // global_addr[31:0]
      g0[3] = (unsigned)((ga >> 32) & 0x01FFFFFFu)  // global_addr[56:32]
            | (2u << 30);                           // type = 2 ("image")
      v8i g1;
      g1[0] = (int)((1u << 16)      // data_size = 1 (2-byte elements)
                  | (1u << 20)      // pad_enable
                  | (4u << 22)      // pad_interval: 32 DWORDs (=64 bf16 row)
                  | (3u << 25));    // pad_amount: 4 DWORDs (=8 bf16)
      g1[1] = (int)((unsigned)(NF & 0xffff) << 16);          // tensor_dim0 lo
      g1[2] = (int)(((NF >> 16) & 0xffff)
                  | ((unsigned)(NPAD & 0xffff) << 16));      // dim0 hi, dim1 lo
      g1[3] = (int)(((NPAD >> 16) & 0xffff)
                  | ((unsigned)TK << 16));                   // dim1 hi, tile_dim0=64
      g1[4] = (int)BM;                                       // tile_dim1=128
      g1[5] = (int)NF;                                       // tensor_dim0_stride=512
      g1[6] = 0;
      g1[7] = 0;
      v4i gz4 = {0, 0, 0, 0};
      v8i gz8 = {0, 0, 0, 0, 0, 0, 0, 0};
      // clang-23 / therock form: (g0, g1, g2, g3, g4, cpol)
      __builtin_amdgcn_tensor_load_to_lds(g0, g1, gz4, gz4, gz8, 0);
    }
#else
    // fallback: manual A staging (buffer is padded, no guards needed)
    for (int i = 0; i < 32; ++i) {
      int e = i * 256 + tid, r = e >> 6, c = e & 63;
      As[r][c] = Abf[(size_t)(row0 + r) * K + kb + c];
    }
#endif
    // stage W tile transposed (f32 -> bf16): Bs[col][k]
    for (int i = 0; i < 16; ++i) {
      int e = i * 256 + tid, k = e >> 6, o = e & 63;
      Bs[o][k] = (bf16_t)Bh[(size_t)(kb + k) * 64 + o];
    }
#if GAT_USE_TDM
    if (wave == 0) __builtin_amdgcn_s_wait_tensorcnt(0);
#endif
    __syncthreads();
    for (int kk = 0; kk < TK; kk += 32) {
      // A frag: row = lane%16, K = kbase..+7 and kbase+16..+23, kbase=(lane/16)*8
      const bf16_t* ap = &As[wave * 16 + lrow][kk + lhalf * 8];
      v16bf afrag = ld_frag(ap, ap + 16);
      for (int t = 0; t < 4; ++t) {
        // B frag: col = lane%16, K = kbase..+7 and kbase+16..+23, kbase=(lane/16)*8
        const bf16_t* bp = &Bs[t * 16 + lrow][kk + lhalf * 8];
        v16bf bfrag = ld_frag(bp, bp + 16);
        acc[t] = wmma_bf16(afrag, bfrag, acc[t]);
      }
    }
  }
  // C layout: VGPR r -> row = r + 8*(lane>=16), col = lane%16 (+16*t)
  const int rbase = row0 + wave * 16 + lhalf * 8;
  for (int t = 0; t < 4; ++t) {
    int col = t * 16 + lrow;
    for (int r = 0; r < 8; ++r)
      outT[(size_t)col * NPAD + rbase + r] = (bf16_t)acc[t][r];
  }
}

// =====================================================================
// Kernel 3: f1/f2 = Wh . a  (reads transposed bf16 Wh, coalesced in n)
// =====================================================================
template<int NH>
__global__ void f1f2_kernel(const bf16_t* __restrict__ WhT,
                            const float* __restrict__ avec,
                            float* __restrict__ f1, float* __restrict__ f2) {
  int idx = blockIdx.x * blockDim.x + threadIdx.x;
  if (idx >= NH * NPAD) return;
  int h = idx / NPAD, n = idx % NPAD;
  const bf16_t* base = WhT + (size_t)h * 64 * NPAD + n;
  const float* a1 = avec + h * 2 * NHID;
  float s1 = 0.f, s2 = 0.f;
  for (int o = 0; o < NHID; ++o) {
    float w = (float)base[(size_t)o * NPAD];
    s1 += w * a1[o];
    s2 += w * a1[NHID + o];
  }
  f1[idx] = s1;
  f2[idx] = s2;
}

// =====================================================================
// Kernel 4: row max. lrelu is monotone => max_j e = lrelu(f1_i + max f2_j)
// Scans only the bitmask (3.2 MB), not adj.
// =====================================================================
template<int NH>
__global__ __launch_bounds__(256) void rowmax_kernel(
    const unsigned* __restrict__ mask, const float* __restrict__ f1,
    const float* __restrict__ f2, float* __restrict__ m) {
  const int i = blockIdx.x;          // < NN
  const int tid = threadIdx.x;
  float best[NH];
  for (int h = 0; h < NH; ++h) best[h] = -3.0e38f;
  const unsigned* mrow = mask + (size_t)i * WPR;
  for (int wd = tid; wd < WPR; wd += 256) {
    unsigned bits = mrow[wd];
    while (bits) {
      int b = __builtin_ctz(bits); bits &= bits - 1;
      int j = wd * 32 + b;
      for (int h = 0; h < NH; ++h)
        best[h] = fmaxf(best[h], f2[(size_t)h * NPAD + j]);
    }
  }
  __shared__ float red[NH * 256];
  for (int h = 0; h < NH; ++h) red[h * 256 + tid] = best[h];
  __syncthreads();
  for (int s = 128; s > 0; s >>= 1) {
    if (tid < s)
      for (int h = 0; h < NH; ++h)
        red[h * 256 + tid] = fmaxf(red[h * 256 + tid], red[h * 256 + tid + s]);
    __syncthreads();
  }
  if (tid < NH) {
    float v = f1[(size_t)tid * NPAD + i] + red[tid * 256];
    m[(size_t)tid * NPAD + i] = v > 0.f ? v : GAT_ALPHA * v;
  }
}

// =====================================================================
// Kernel 5: layer-1 attention, flash-style. Block = 8 waves = 8 heads
// sharing 32 rows. P tile (16x32 bf16) built in registers from mask bits
// + exp, used directly as the WMMA A fragment. All 4 B fragments loaded
// before the 8 WMMAs so the 128B loads pipeline under one wait.
// =====================================================================
__global__ __launch_bounds__(256) void attn1_kernel(
    const unsigned* __restrict__ mask, const float* __restrict__ f1,
    const float* __restrict__ f2, const float* __restrict__ m,
    const bf16_t* __restrict__ WhT, bf16_t* __restrict__ h1) {
  const int row0 = blockIdx.x * 32;
  const int h = threadIdx.x >> 5;          // wave == head
  const int lane = threadIdx.x & 31;
  const int lrow = lane & 15, lhalf = lane >> 4, kb = lhalf * 8;
  const float* f2h = f2 + (size_t)h * NPAD;
  const bf16_t* WhTh = WhT + (size_t)h * 64 * NPAD;

  float mi[2], f1i[2];
  int rg[2];
  for (int mt = 0; mt < 2; ++mt) {
    rg[mt]  = row0 + mt * 16 + lrow;
    mi[mt]  = m[(size_t)h * NPAD + rg[mt]];
    f1i[mt] = f1[(size_t)h * NPAD + rg[mt]];
  }
  v8f acc[2][4];
  for (int mt = 0; mt < 2; ++mt)
    for (int t = 0; t < 4; ++t) acc[mt][t] = zero8();
  float dsum[2] = {0.f, 0.f};

  for (int j0 = 0; j0 < NPAD; j0 += 32) {
    const int wword = j0 >> 5;
    unsigned bits[2];
    bits[0] = mask[(size_t)rg[0] * WPR + wword];
    bits[1] = mask[(size_t)rg[1] * WPR + wword];
    if (ballot32((bits[0] | bits[1]) != 0) == 0) continue;  // empty chunk
    v16bf pa[2];
    for (int mt = 0; mt < 2; ++mt) {
      bf16_t* ae = reinterpret_cast<bf16_t*>(&pa[mt]);
      for (int idx = 0; idx < 16; ++idx) {
        int k = lhalf * 16 + idx;   // A-frag K order
        float p = 0.f;
        if (bits[mt] & (1u << k)) {
          float e = f1i[mt] + f2h[j0 + k];
          e = e > 0.f ? e : GAT_ALPHA * e;
          p = __expf(e - mi[mt]);
          dsum[mt] += p;
        }
        ae[idx] = (bf16_t)p;
      }
    }
    v16bf bf4[4];
    for (int t = 0; t < 4; ++t) {
      const bf16_t* bp = WhTh + (size_t)(t * 16 + lrow) * NPAD + j0 + lhalf * 16;
      bf4[t] = ld_frag(bp, bp + 8);
    }
    for (int t = 0; t < 4; ++t) {
      acc[0][t] = wmma_bf16(pa[0], bf4[t], acc[0][t]);
      acc[1][t] = wmma_bf16(pa[1], bf4[t], acc[1][t]);
    }
  }
  // epilogue: softmax normalize + ELU + concat store (bf16 for next GEMM)
  for (int mt = 0; mt < 2; ++mt) {
    float dfull = dsum[mt] + __shfl_xor(dsum[mt], 16, 32);  // row lane%16 denom
    for (int r = 0; r < 8; ++r) {
      float dr = __shfl(dfull, r + lhalf * 8, 32);          // denom of C-row
      float sc = dr > 0.f ? 1.0f / dr : 0.f;
      int rowg = row0 + mt * 16 + r + lhalf * 8;
      for (int t = 0; t < 4; ++t) {
        float v = acc[mt][t][r] * sc;
        v = v > 0.f ? v : (__expf(v) - 1.0f);               // ELU
        h1[(size_t)rowg * (NHEADS * NHID) + h * NHID + t * 16 + lrow] = (bf16_t)v;
      }
    }
  }
}

// =====================================================================
// Kernel 6: layer-2 attention (single head). 8 waves split the j range
// (interleaved 32-wide chunks), partial numerators/denominators combined
// through 64 KB of LDS (WGP has 320 KB).
// =====================================================================
__global__ __launch_bounds__(256) void attn2_kernel(
    const unsigned* __restrict__ mask, const float* __restrict__ g1,
    const float* __restrict__ g2, const float* __restrict__ m2,
    const bf16_t* __restrict__ Wh2T, float* __restrict__ h2) {
  const int row0 = blockIdx.x * 32;
  const int w = threadIdx.x >> 5;
  const int lane = threadIdx.x & 31;
  const int lrow = lane & 15, lhalf = lane >> 4, kb = lhalf * 8;

  float mi[2], f1i[2];
  int rg[2];
  for (int mt = 0; mt < 2; ++mt) {
    rg[mt]  = row0 + mt * 16 + lrow;
    mi[mt]  = m2[rg[mt]];
    f1i[mt] = g1[rg[mt]];
  }
  v8f acc[2][4];
  for (int mt = 0; mt < 2; ++mt)
    for (int t = 0; t < 4; ++t) acc[mt][t] = zero8();
  float dsum[2] = {0.f, 0.f};

  for (int c = w; c < NPAD / 32; c += 8) {     // j-split across waves
    const int j0 = c * 32;
    unsigned bits[2];
    bits[0] = mask[(size_t)rg[0] * WPR + c];
    bits[1] = mask[(size_t)rg[1] * WPR + c];
    if (ballot32((bits[0] | bits[1]) != 0) == 0) continue;
    v16bf pa[2];
    for (int mt = 0; mt < 2; ++mt) {
      bf16_t* ae = reinterpret_cast<bf16_t*>(&pa[mt]);
      for (int idx = 0; idx < 16; ++idx) {
        int k = lhalf * 16 + idx;
        float p = 0.f;
        if (bits[mt] & (1u << k)) {
          float e = f1i[mt] + g2[j0 + k];
          e = e > 0.f ? e : GAT_ALPHA * e;
          p = __expf(e - mi[mt]);
          dsum[mt] += p;
        }
        ae[idx] = (bf16_t)p;
      }
    }
    v16bf bf4[4];
    for (int t = 0; t < 4; ++t) {
      const bf16_t* bp = Wh2T + (size_t)(t * 16 + lrow) * NPAD + j0 + lhalf * 16;
      bf4[t] = ld_frag(bp, bp + 8);
    }
    for (int t = 0; t < 4; ++t) {
      acc[0][t] = wmma_bf16(pa[0], bf4[t], acc[0][t]);
      acc[1][t] = wmma_bf16(pa[1], bf4[t], acc[1][t]);
    }
  }
  __shared__ float accl[8][2][4][8][32];   // 64 KB partial numerators
  __shared__ float denl[8][2][16];
  for (int mt = 0; mt < 2; ++mt) {
    float dfull = dsum[mt] + __shfl_xor(dsum[mt], 16, 32);
    if (lane < 16) denl[w][mt][lane] = dfull;
    for (int t = 0; t < 4; ++t)
      for (int r = 0; r < 8; ++r) accl[w][mt][t][r][lane] = acc[mt][t][r];
  }
  __syncthreads();
  {
    const int mt = w >> 2, t = w & 3;      // each wave reduces one (mt,t) slice
    for (int r = 0; r < 8; ++r) {
      float s = 0.f;
      for (int ww = 0; ww < 8; ++ww) s += accl[ww][mt][t][r][lane];
      int rowr = r + lhalf * 8;
      float d = 0.f;
      for (int ww = 0; ww < 8; ++ww) d += denl[ww][mt][rowr];
      float sc = d > 0.f ? 1.0f / d : 0.f;
      h2[(size_t)(row0 + mt * 16 + rowr) * NHID + t * 16 + lrow] = s * sc;
    }
  }
}

// =====================================================================
// MLP head (tiny: 0.2 GFLOP total) — plain f32 VALU
// =====================================================================
__global__ void mlp1_kernel(const float* __restrict__ h2,
                            const float* __restrict__ w1,
                            const float* __restrict__ b1,
                            float* __restrict__ h3) {
  int idx = blockIdx.x * blockDim.x + threadIdx.x;
  if (idx >= NN * 200) return;
  int n = idx / 200, q = idx % 200;
  float s = b1[q];
  for (int k = 0; k < NHID; ++k) s += h2[(size_t)n * NHID + k] * w1[k * 200 + q];
  h3[idx] = s > 0.f ? s : (__expf(s) - 1.0f);
}

__global__ void mlp2_kernel(const float* __restrict__ h3,
                            const float* __restrict__ w2,
                            const float* __restrict__ b2,
                            float* __restrict__ out) {
  int idx = blockIdx.x * blockDim.x + threadIdx.x;
  if (idx >= NN * NCLASS) return;
  int n = idx / NCLASS, c = idx % NCLASS;
  float s = b2[c];
  for (int q = 0; q < 200; ++q) s += h3[(size_t)n * 200 + q] * w2[q * NCLASS + c];
  out[idx] = s;
}

// =====================================================================
extern "C" void kernel_launch(void* const* d_in, const int* in_sizes, int n_in,
                              void* d_out, int out_size, void* d_ws, size_t ws_size,
                              hipStream_t stream) {
  (void)in_sizes; (void)n_in; (void)out_size; (void)ws_size;
  const float* x     = (const float*)d_in[0];   // [5000,512]
  const float* adj   = (const float*)d_in[1];   // [5000,5000]
  const float* W_h   = (const float*)d_in[2];   // [8,512,64]
  const float* a_h   = (const float*)d_in[3];   // [8,128]
  const float* W_o   = (const float*)d_in[4];   // [512,64]
  const float* a_o   = (const float*)d_in[5];   // [128]
  const float* fc1_w = (const float*)d_in[6];   // [64,200]
  const float* fc1_b = (const float*)d_in[7];   // [200]
  const float* fc2_w = (const float*)d_in[8];   // [200,40]
  const float* fc2_b = (const float*)d_in[9];   // [40]
  float* out = (float*)d_out;                   // [5000,40]

  // ---- workspace carve (~25 MB) ----
  char* base = (char*)d_ws;
  size_t off = 0;
  auto carve = [&](size_t bytes) {
    void* p = base + off;
    off = (off + bytes + 255) & ~(size_t)255;
    return p;
  };
  unsigned* mask = (unsigned*)carve((size_t)NPAD * WPR * 4);          // 3.3 MB
  bf16_t* xb     = (bf16_t*)carve((size_t)NPAD * NF * 2);             // 5.2 MB
  bf16_t* WhT    = (bf16_t*)carve((size_t)NHEADS * NHID * NPAD * 2);  // 5.2 MB
  float*  f1     = (float*) carve((size_t)NHEADS * NPAD * 4);
  float*  f2     = (float*) carve((size_t)NHEADS * NPAD * 4);
  float*  m1     = (float*) carve((size_t)NHEADS * NPAD * 4);
  bf16_t* h1     = (bf16_t*)carve((size_t)NPAD * NHEADS * NHID * 2);  // 5.2 MB
  bf16_t* Wh2T   = (bf16_t*)carve((size_t)NHID * NPAD * 2);
  float*  g1     = (float*) carve((size_t)NPAD * 4);
  float*  g2     = (float*) carve((size_t)NPAD * 4);
  float*  m2     = (float*) carve((size_t)NPAD * 4);
  float*  h2     = (float*) carve((size_t)NPAD * NHID * 4);
  float*  h3     = (float*) carve((size_t)NN * 200 * 4);

  // 0) x -> padded bf16 (TDM source tensor)
  tobf16_kernel<<<(NPAD * NF + 255) / 256, 256, 0, stream>>>(x, xb);
  // 1) adjacency -> bitmask (the only full read of the 100 MB adj)
  mask_kernel<<<NPAD, 256, 0, stream>>>(adj, mask);
  // 2) per-head feature GEMM (bf16 WMMA, TDM-staged A tiles)
  gemm512_kernel<<<dim3(NPAD / 128, NHEADS), 256, 0, stream>>>(xb, W_h, WhT);
  // 3) attention logits pieces
  f1f2_kernel<NHEADS><<<(NHEADS * NPAD + 255) / 256, 256, 0, stream>>>(
      WhT, a_h, f1, f2);
  // 4) analytic row maxima from bitmask
  rowmax_kernel<NHEADS><<<NN, 256, 0, stream>>>(mask, f1, f2, m1);
  // 5) layer-1 masked softmax + aggregation (WMMA) + ELU + concat
  attn1_kernel<<<NPAD / 32, 256, 0, stream>>>(mask, f1, f2, m1, WhT, h1);
  // 6) layer-2 GEMM h1 @ W_o (bf16 WMMA, TDM-staged A tiles)
  gemm512_kernel<<<dim3(NPAD / 128, 1), 256, 0, stream>>>(h1, W_o, Wh2T);
  // 7) layer-2 logits + row maxima
  f1f2_kernel<1><<<(NPAD + 255) / 256, 256, 0, stream>>>(Wh2T, a_o, g1, g2);
  rowmax_kernel<1><<<NN, 256, 0, stream>>>(mask, g1, g2, m2);
  // 8) layer-2 attention (WMMA, j-split + LDS combine)
  attn2_kernel<<<NPAD / 32, 256, 0, stream>>>(mask, g1, g2, m2, Wh2T, h2);
  // 9) MLP head
  mlp1_kernel<<<(NN * 200 + 255) / 256, 256, 0, stream>>>(h2, fc1_w, fc1_b, h3);
  mlp2_kernel<<<(NN * NCLASS + 255) / 256, 256, 0, stream>>>(h3, fc2_w, fc2_b, out);
}
